// MultiHeadAttentionConv_61117384622153
// MI455X (gfx1250) — compile-verified, matrix operand order checked by reading, not run on hardware
//
#include <hip/hip_runtime.h>
#include <hip/hip_bf16.h>

typedef __attribute__((ext_vector_type(16))) _Float16 v16h;
typedef __attribute__((ext_vector_type(8)))  _Float16 v8h;
typedef __attribute__((ext_vector_type(8)))  float    v8f;
typedef __attribute__((ext_vector_type(4)))  float    f32x4;

// Problem constants
#define NTOK   2048          // sequence length N
#define NHEAD  8
#define DK     512
#define CCH    256           // C
#define POS    64            // H*W
#define OCH    64            // OUT_CH_K == OUT_CH_V

static __device__ __forceinline__ v8f wmma_f16f32(v16h a, v16h b, v8f c) {
  // D = A(16x32 f16) * B(32x16 f16) + C(16x16 f32)
  return __builtin_amdgcn_wmma_f32_16x16x32_f16(false, a, false, b, (short)0, c,
                                                false, false);
}

// Load one 16x32 (f16) fragment from a row-major matrix R (leading dim ld).
// Lane l<16 holds row (rowbase+l), K = {k0..k0+7, k0+16..k0+23};
// lane l>=16 holds row (rowbase+l-16), K = {k0+8..k0+15, k0+24..k0+31}.
static __device__ __forceinline__ v16h frag_rowmajor_f16(
    const _Float16* __restrict__ R, int ld, int row, int k0, int lane) {
  const int hi = lane >> 4;
  const _Float16* p = R + (size_t)row * ld + (k0 + hi * 8);
  v8h c0 = *(const v8h*)p;
  v8h c1 = *(const v8h*)(p + 16);
  v16h f;
#pragma unroll
  for (int i = 0; i < 8; ++i) { f[i] = c0[i]; f[8 + i] = c1[i]; }
  return f;
}

// ---------------------------------------------------------------------------
// K0: convert projection weights fp32 -> f16
// ---------------------------------------------------------------------------
__global__ void cvt_weights_kernel(const float* __restrict__ wq,
                                   const float* __restrict__ wk,
                                   const float* __restrict__ wv,
                                   _Float16* __restrict__ oq,
                                   _Float16* __restrict__ ok,
                                   _Float16* __restrict__ ov) {
  int i = blockIdx.x * blockDim.x + threadIdx.x;
  if (i < OCH * CCH) {
    oq[i] = (_Float16)wq[i];
    ok[i] = (_Float16)wk[i];
    ov[i] = (_Float16)wv[i];
  }
}

// ---------------------------------------------------------------------------
// K1: 1x1-conv projection: y_n(64x64) = W(64x256) @ x_n(256x64).
// One wave does a whole sample (4x4 WMMA macro-tile, 16 accumulators);
// W fragments are reused across all 4 position tiles.
// Output written directly into attention layout:
//   vmode==0:  dst[head][n][d]   (Q,K : (h,N,dk) f16)
//   vmode==1:  dst[head][d][n]   (V transposed : (h,dv,N) f16)
// ---------------------------------------------------------------------------
__global__ void __launch_bounds__(256)
proj_kernel(const float* __restrict__ x, const _Float16* __restrict__ W16,
            const float* __restrict__ bias, _Float16* __restrict__ dst,
            int vmode) {
  const int lane = threadIdx.x & 31;
  const int wave = threadIdx.x >> 5;
  const int n    = blockIdx.x * 8 + wave;        // sample = global wave id
  const int nl   = lane & 15;
  const int hi   = lane >> 4;
  const float* xn = x + (size_t)n * (CCH * POS); // (c, p) row-major, ld=64

  v8f acc[4][4];
#pragma unroll
  for (int i = 0; i < 4; ++i)
#pragma unroll
    for (int j = 0; j < 4; ++j) acc[i][j] = {};

  for (int k0 = 0; k0 < CCH; k0 += 32) {
    v16h a[4];
#pragma unroll
    for (int i = 0; i < 4; ++i)
      a[i] = frag_rowmajor_f16(W16, CCH, i * 16 + nl, k0, lane);
    const int kb = k0 + hi * 8;
#pragma unroll
    for (int j = 0; j < 4; ++j) {
      const int p = j * 16 + nl;                 // spatial position
      v16h b;                                    // B[k=c][n=p] = x[c][p]
#pragma unroll
      for (int t = 0; t < 8; ++t) {
        b[t]     = (_Float16)xn[(kb + t) * POS + p];
        b[8 + t] = (_Float16)xn[(kb + 16 + t) * POS + p];
      }
#pragma unroll
      for (int i = 0; i < 4; ++i) acc[i][j] = wmma_f16f32(a[i], b, acc[i][j]);
    }
  }

#pragma unroll
  for (int i = 0; i < 4; ++i)
#pragma unroll
    for (int j = 0; j < 4; ++j)
#pragma unroll
      for (int r = 0; r < 8; ++r) {
        const int o = i * 16 + r + hi * 8;       // out channel 0..63
        const int p = j * 16 + nl;               // position 0..63
        const float val = acc[i][j][r] + bias[o];
        const int head = o >> 3;
        const int d    = (o & 7) * POS + p;      // 0..511
        if (vmode)
          dst[(size_t)(head * DK + d) * NTOK + n] = (_Float16)val;
        else
          dst[(size_t)(head * NTOK + n) * DK + d] = (_Float16)val;
      }
}

// ---------------------------------------------------------------------------
// K2: raw scores S = (Q @ K^T) / sqrt(512) into the attn output region (fp32)
// One wave per 64x64 macro-tile (16 accumulators); per K-step 4 A-frags +
// 4 B-frags feed 16 WMMAs. 8 waves per block.
// ---------------------------------------------------------------------------
__global__ void __launch_bounds__(256)
scores_kernel(const _Float16* __restrict__ Qh, const _Float16* __restrict__ Kh,
              float* __restrict__ attn) {
  const int lane = threadIdx.x & 31;
  const int wave = threadIdx.x >> 5;
  const int wg   = blockIdx.x * 8 + wave;
  const int h    = wg >> 10;                     // 32*32 macro tiles per head
  const int rem  = wg & 1023;
  const int qb   = (rem >> 5) * 64;              // query-row base
  const int kb0  = (rem & 31) * 64;              // key-col base
  const int nl   = lane & 15;
  const int hi   = lane >> 4;
  const _Float16* Q = Qh + (size_t)h * NTOK * DK;
  const _Float16* K = Kh + (size_t)h * NTOK * DK;

  v8f acc[4][4];
#pragma unroll
  for (int i = 0; i < 4; ++i)
#pragma unroll
    for (int j = 0; j < 4; ++j) acc[i][j] = {};

  for (int k0 = 0; k0 < DK; k0 += 32) {
    __builtin_prefetch(Q + (size_t)(qb + nl) * DK + k0 + 32, 0, 1);
    __builtin_prefetch(K + (size_t)(kb0 + nl) * DK + k0 + 32, 0, 1);
    v16h a[4];
#pragma unroll
    for (int i = 0; i < 4; ++i)
      a[i] = frag_rowmajor_f16(Q, DK, qb + i * 16 + nl, k0, lane);
#pragma unroll
    for (int j = 0; j < 4; ++j) {
      // B[k=d][n=key] = Kh[key][d] : same row-major gather over Kh rows
      v16h b = frag_rowmajor_f16(K, DK, kb0 + j * 16 + nl, k0, lane);
#pragma unroll
      for (int i = 0; i < 4; ++i) acc[i][j] = wmma_f16f32(a[i], b, acc[i][j]);
    }
  }

  float* S = attn + (size_t)h * NTOK * NTOK;
  const float scale = 0.04419417382415922f;      // 1/sqrt(512)
#pragma unroll
  for (int i = 0; i < 4; ++i)
#pragma unroll
    for (int r = 0; r < 8; ++r) {
      const int row = qb + i * 16 + r + hi * 8;
#pragma unroll
      for (int j = 0; j < 4; ++j)
        S[(size_t)row * NTOK + kb0 + j * 16 + nl] = acc[i][j][r] * scale;
    }
}

// ---------------------------------------------------------------------------
// K3: row softmax in place over the 2048-long key axis. One block per row.
// ---------------------------------------------------------------------------
__global__ void __launch_bounds__(256)
softmax_kernel(float* __restrict__ attn) {
  float* row = attn + (size_t)blockIdx.x * NTOK;
  const int t = threadIdx.x;
  __shared__ float red[256];

  float v[8];
  float m = -3.4e38f;
#pragma unroll
  for (int i = 0; i < 8; ++i) {
    v[i] = row[t + 256 * i];
    m = fmaxf(m, v[i]);
  }
  red[t] = m;
  __syncthreads();
  for (int s = 128; s > 0; s >>= 1) {
    if (t < s) red[t] = fmaxf(red[t], red[t + s]);
    __syncthreads();
  }
  const float rmax = red[0];
  __syncthreads();

  float sum = 0.f;
#pragma unroll
  for (int i = 0; i < 8; ++i) {
    v[i] = __expf(v[i] - rmax);
    sum += v[i];
  }
  red[t] = sum;
  __syncthreads();
  for (int s = 128; s > 0; s >>= 1) {
    if (t < s) red[t] += red[t + s];
    __syncthreads();
  }
  const float inv = 1.0f / red[0];
#pragma unroll
  for (int i = 0; i < 8; ++i) row[t + 256 * i] = v[i] * inv;
}

// ---------------------------------------------------------------------------
// K4: O = attn @ V.  A = fp32 attn (converted to f16 in-register),
// B = Vt (h, d, N) f16 so the B-fragment load is contiguous.
// One wave per 64x32 macro-tile (8 accumulators); per K-step 4 A-frags +
// 2 B-frags feed 8 WMMAs.
// ---------------------------------------------------------------------------
__global__ void __launch_bounds__(256)
av_kernel(const float* __restrict__ attn, const _Float16* __restrict__ Vt,
          float* __restrict__ O) {
  const int lane = threadIdx.x & 31;
  const int wave = threadIdx.x >> 5;
  const int wg   = blockIdx.x * 8 + wave;
  const int h    = wg >> 9;                      // 32*16 macro tiles per head
  const int rem  = wg & 511;
  const int qb   = (rem >> 4) * 64;              // query-row base
  const int db   = (rem & 15) * 32;              // d-col base
  const int nl   = lane & 15;
  const int hi   = lane >> 4;
  const float*    A = attn + (size_t)h * NTOK * NTOK;
  const _Float16* V = Vt   + (size_t)h * DK * NTOK;

  v8f acc[4][2];
#pragma unroll
  for (int i = 0; i < 4; ++i)
#pragma unroll
    for (int j = 0; j < 2; ++j) acc[i][j] = {};

  for (int k0 = 0; k0 < NTOK; k0 += 32) {
    __builtin_prefetch(A + (size_t)(qb + nl) * NTOK + k0 + 32, 0, 1);
    const int kb = k0 + hi * 8;
    v16h a[4];
#pragma unroll
    for (int i = 0; i < 4; ++i) {
      const float* ap = A + (size_t)(qb + i * 16 + nl) * NTOK + kb;
      f32x4 x0 = *(const f32x4*)(ap);
      f32x4 x1 = *(const f32x4*)(ap + 4);
      f32x4 x2 = *(const f32x4*)(ap + 16);
      f32x4 x3 = *(const f32x4*)(ap + 20);
#pragma unroll
      for (int t = 0; t < 4; ++t) {
        a[i][t]      = (_Float16)x0[t];
        a[i][4 + t]  = (_Float16)x1[t];
        a[i][8 + t]  = (_Float16)x2[t];
        a[i][12 + t] = (_Float16)x3[t];
      }
    }
#pragma unroll
    for (int j = 0; j < 2; ++j) {
      // B[k=key][n=d] = Vt[d][key] : contiguous over keys
      v16h b = frag_rowmajor_f16(V, NTOK, db + j * 16 + nl, k0, lane);
#pragma unroll
      for (int i = 0; i < 4; ++i) acc[i][j] = wmma_f16f32(a[i], b, acc[i][j]);
    }
  }

#pragma unroll
  for (int i = 0; i < 4; ++i)
#pragma unroll
    for (int r = 0; r < 8; ++r) {
      const int qrow = qb + i * 16 + r + hi * 8;
#pragma unroll
      for (int j = 0; j < 2; ++j)
        O[((size_t)h * NTOK + qrow) * DK + db + j * 16 + nl] = acc[i][j][r];
    }
}

// ---------------------------------------------------------------------------
// K5: fc (1x1 conv 256<-64) + bias + residual + InstanceNorm per (n,c).
// One block per sample; thread t owns output channel oc=t (all 64 positions).
// ---------------------------------------------------------------------------
__global__ void __launch_bounds__(256)
fc_norm_kernel(const float* __restrict__ O, const float* __restrict__ Wfc,
               const float* __restrict__ bfc, const float* __restrict__ qres,
               float* __restrict__ out) {
  __shared__ float sO[OCH * POS];                // 16 KB: (c64, p)
  const int n = blockIdx.x;
  const int t = threadIdx.x;

  // Stage attention output for this sample, undoing (h,N,d) -> (N,64,8,8)
#pragma unroll
  for (int i = 0; i < 16; ++i) {
    const int idx = t * 16 + i;
    const int c64 = idx >> 6;
    const int p   = idx & 63;
    sO[idx] = O[((size_t)(c64 >> 3) * NTOK + n) * DK + (c64 & 7) * POS + p];
  }
  __syncthreads();

  const int oc = t;
  float y[POS];
#pragma unroll
  for (int p = 0; p < POS; ++p) y[p] = 0.f;

  const f32x4* s4 = (const f32x4*)sO;
  for (int c = 0; c < OCH; ++c) {
    const float w = Wfc[oc * OCH + c];
#pragma unroll
    for (int p4 = 0; p4 < 16; ++p4) {
      f32x4 sv = s4[c * 16 + p4];                // LDS broadcast
      y[p4 * 4 + 0] += w * sv[0];
      y[p4 * 4 + 1] += w * sv[1];
      y[p4 * 4 + 2] += w * sv[2];
      y[p4 * 4 + 3] += w * sv[3];
    }
  }

  const float b = bfc[oc];
  const float* qr = qres + ((size_t)n * CCH + oc) * POS;
  float sum = 0.f, sq = 0.f;
#pragma unroll
  for (int p = 0; p < POS; ++p) {
    y[p] += b + qr[p];
    sum += y[p];
    sq  += y[p] * y[p];
  }
  const float mean = sum * (1.f / 64.f);
  const float var  = sq * (1.f / 64.f) - mean * mean;   // population var
  const float inv  = rsqrtf(var + 1e-5f);
  float* op = out + ((size_t)n * CCH + oc) * POS;
#pragma unroll
  for (int p = 0; p < POS; ++p) op[p] = (y[p] - mean) * inv;
}

// ---------------------------------------------------------------------------
extern "C" void kernel_launch(void* const* d_in, const int* in_sizes, int n_in,
                              void* d_out, int out_size, void* d_ws,
                              size_t ws_size, hipStream_t stream) {
  (void)in_sizes; (void)n_in; (void)out_size; (void)ws_size;
  const float* q   = (const float*)d_in[0];
  const float* k   = (const float*)d_in[1];
  const float* v   = (const float*)d_in[2];
  const float* Wq  = (const float*)d_in[3];
  const float* bq  = (const float*)d_in[4];
  const float* Wk  = (const float*)d_in[5];
  const float* bk  = (const float*)d_in[6];
  const float* Wv  = (const float*)d_in[7];
  const float* bv  = (const float*)d_in[8];
  const float* Wfc = (const float*)d_in[9];
  const float* bfc = (const float*)d_in[10];

  float* out  = (float*)d_out;                         // (N,256,8,8)
  float* attn = out + (size_t)NTOK * CCH * POS;        // (8,2048,2048)

  // Workspace carve-up (halfs): weights, Qh, Kh, Vt(f16), then O (fp32)
  _Float16* hws  = (_Float16*)d_ws;
  _Float16* Wq16 = hws;                                // 16384 halfs
  _Float16* Wk16 = Wq16 + OCH * CCH;
  _Float16* Wv16 = Wk16 + OCH * CCH;
  _Float16* Qh   = hws + 49152;                        // 16B-aligned
  _Float16* Kh   = Qh + (size_t)NHEAD * NTOK * DK;     // 8.39M halfs each
  _Float16* Vt   = Kh + (size_t)NHEAD * NTOK * DK;
  float*    O    = (float*)(Vt + (size_t)NHEAD * NTOK * DK);

  cvt_weights_kernel<<<64, 256, 0, stream>>>(Wq, Wk, Wv, Wq16, Wk16, Wv16);

  proj_kernel<<<256, 256, 0, stream>>>(q, Wq16, bq, Qh, 0);
  proj_kernel<<<256, 256, 0, stream>>>(k, Wk16, bk, Kh, 0);
  proj_kernel<<<256, 256, 0, stream>>>(v, Wv16, bv, Vt, 1);

  scores_kernel<<<1024, 256, 0, stream>>>(Qh, Kh, attn);
  softmax_kernel<<<16384, 256, 0, stream>>>(attn);
  av_kernel<<<512, 256, 0, stream>>>(attn, Vt, O);
  fc_norm_kernel<<<2048, 256, 0, stream>>>(O, Wfc, bfc, q, out);
}
